// NequIP_75453985456895
// MI455X (gfx1250) — compile-verified, matrix operand order checked by reading, not run on hardware
//
#include <hip/hip_runtime.h>
#include <math.h>

// ---------------------------------------------------------------------------
// NequIP-style 3-layer equivariant GNN for MI455X (gfx1250, wave32, WMMA).
// Radial MLP GEMMs run on v_wmma_f32_16x16x32_f16 (f16 in, f32 accumulate).
// Everything else (gather/scatter/mix/gate) is L2-resident memory work.
// ---------------------------------------------------------------------------

typedef _Float16 f16;
typedef __attribute__((ext_vector_type(16))) _Float16 v16h;
typedef __attribute__((ext_vector_type(8)))  float    v8f;

#define NN      16384
#define EE      262144
#define CC      32
#define NBQ     8
#define HH      64
#define MMC     9
#define NLAYERS 3
#define W2OUT   192          // C * 2 * (LMAX+1)
#define RMAXF   5.0f
#define INV_AVG (1.0f/16.0f)
#define PI_F    3.14159265358979f

// A/B fragment K mapping for 16-bit 16x32 WMMA operands (ISA 7.12.2):
// lanes 0-15 (half=0): h[0..7]->K0..7,  h[8..15]->K16..23
// lanes16-31 (half=1): h[0..7]->K8..15, h[8..15]->K24..31
__device__ __forceinline__ int kmap(int j, int half) {
    return (j < 8) ? (half * 8 + j) : (16 + half * 8 + (j - 8));
}

// ---------------------------------------------------------------------------
// Edge geometry: Y [E,9] f32, bessel [E,8] f16 (packed for A-fragment loads),
// cutoff [E] f32. Computed once; shared by all 3 layers.
// ---------------------------------------------------------------------------
__global__ __launch_bounds__(256)
void geom_kernel(const float* __restrict__ pos, const int* __restrict__ snd,
                 const int* __restrict__ rcv, float* __restrict__ Y,
                 float* __restrict__ cut, f16* __restrict__ bess)
{
    int e = blockIdx.x * blockDim.x + threadIdx.x;
    int s = snd[e], r = rcv[e];
    float dx = (pos[r*3+0] - pos[s*3+0]) * (1.0f / RMAXF);
    float dy = (pos[r*3+1] - pos[s*3+1]) * (1.0f / RMAXF);
    float dz = (pos[r*3+2] - pos[s*3+2]) * (1.0f / RMAXF);
    float rr  = sqrtf(dx*dx + dy*dy + dz*dz);
    float inv = 1.0f / (rr + 1e-9f);
    float x = dx*inv, y = dy*inv, z = dz*inv;
    const float s3 = 1.7320508075688772f;

    float* yv = Y + (size_t)e * MMC;
    yv[0] = 1.0f;
    yv[1] = x;  yv[2] = y;  yv[3] = z;
    yv[4] = s3 * x * y;
    yv[5] = s3 * y * z;
    yv[6] = 0.5f * (3.0f * z * z - 1.0f);
    yv[7] = s3 * x * z;
    yv[8] = 0.5f * s3 * (x * x - y * y);

    f16* bp = bess + (size_t)e * NBQ;
#pragma unroll
    for (int n = 1; n <= NBQ; ++n)
        bp[n-1] = (f16)(__sinf((float)n * PI_F * rr) * inv);

    float rc = fminf(rr, 1.0f);
    cut[e] = (rr < 1.0f) ? 0.5f * (__cosf(PI_F * rc) + 1.0f) : 0.0f;
}

// ---------------------------------------------------------------------------
// Pre-swizzle radial MLP weights into WMMA B-fragment layout (f16).
// w1f: [layer][tile(4)][lane(32)][j(16)]  — K padded 8->32 with zeros
// w2f: [layer][tile(12)][kfrag(2)][lane(32)][j(16)]
// ---------------------------------------------------------------------------
__global__ __launch_bounds__(256)
void conv_w1(const float* __restrict__ w1, f16* __restrict__ w1f)
{
    int idx   = blockIdx.x * blockDim.x + threadIdx.x;
    int j     = idx & 15;
    int lane  = (idx >> 4) & 31;
    int t     = (idx >> 9) & 3;
    int layer = idx >> 11;
    int k     = kmap(j, lane >> 4);
    int col   = t * 16 + (lane & 15);
    f16 v = (f16)0.0f;
    if (k < NBQ) v = (f16)w1[((size_t)layer * NBQ + k) * HH + col];
    w1f[idx] = v;
}

__global__ __launch_bounds__(256)
void conv_w2(const float* __restrict__ w2, f16* __restrict__ w2f)
{
    int idx   = blockIdx.x * blockDim.x + threadIdx.x;
    int j     = idx & 15;
    int lane  = (idx >> 4) & 31;
    int kf    = (idx >> 9) & 1;
    int tl    = idx >> 10;          // layer*12 + tile
    int t     = tl % 12;
    int layer = tl / 12;
    int k     = kf * 32 + kmap(j, lane >> 4);
    int col   = t * 16 + (lane & 15);
    w2f[idx] = (f16)w2[((size_t)layer * HH + k) * W2OUT + col];
}

// ---------------------------------------------------------------------------
// Utility: zero agg, init feats (species embedding into l=0 slot)
// ---------------------------------------------------------------------------
__global__ __launch_bounds__(256)
void zero_kernel(float* __restrict__ p, int n)
{
    int i = blockIdx.x * blockDim.x + threadIdx.x;
    if (i < n) p[i] = 0.0f;
}

__global__ __launch_bounds__(256)
void init_feats(const float* __restrict__ embed, const int* __restrict__ species,
                float* __restrict__ feats)
{
    int idx = blockIdx.x * blockDim.x + threadIdx.x;   // n*CC + c
    int c = idx & 31;
    int n = idx >> 5;
    float e = embed[species[n] * CC + c];
    float* f = feats + (size_t)idx * MMC;
    f[0] = e;
#pragma unroll
    for (int m = 1; m < MMC; ++m) f[m] = 0.0f;
}

// ---------------------------------------------------------------------------
// Fused edge kernel (one per layer):
//   GEMM1: bessel[16,8(->32)] x w1[32,64]  via 4x v_wmma_f32_16x16x32_f16
//   SiLU, stage h to LDS as f16
//   GEMM2: h[16,64] x w2[64,192]           via 24x v_wmma_f32_16x16x32_f16
//   stage w to LDS (f32), then lane=channel message + atomic scatter to agg
// 4 waves / block, 16 edges / wave.
// ---------------------------------------------------------------------------
__global__ __launch_bounds__(128)
void edge_kernel(int layer,
                 const f16*   __restrict__ bessel,
                 const float* __restrict__ Y,
                 const float* __restrict__ cutoff,
                 const float* __restrict__ feats,
                 const f16*   __restrict__ w1f,
                 const f16*   __restrict__ w2f,
                 const float* __restrict__ b1,
                 const float* __restrict__ b2,
                 const int*   __restrict__ snd,
                 const int*   __restrict__ rcv,
                 float*       __restrict__ agg)
{
    __shared__ f16   hstage[4][16][HH];      //  8 KB
    __shared__ float wstage[4][16][W2OUT];   // 48 KB

    const int warp = threadIdx.x >> 5;
    const int lane = threadIdx.x & 31;
    const int half = lane >> 4;
    const int mrow = lane & 15;
    const int tileBase = (blockIdx.x * 4 + warp) * 16;

    // ---- GEMM1 A fragment: 16 edges x K(8, padded to 32) f16 ----
    v16h a1 = {};
    if (half == 0) {
        const f16* bp = bessel + (size_t)(tileBase + mrow) * NBQ;
#pragma unroll
        for (int j = 0; j < 8; ++j) a1[j] = bp[j];
    }

    // ---- GEMM1: h = silu(bessel @ w1 + b1), staged to LDS as f16 ----
#pragma unroll
    for (int t = 0; t < 4; ++t) {
        v16h bf = *(const v16h*)(w1f + (((size_t)layer * 4 + t) * 32 + lane) * 16);
        v8f c = {};
        c = __builtin_amdgcn_wmma_f32_16x16x32_f16(false, a1, false, bf,
                                                   (short)0, c, false, false);
        float bb = b1[layer * HH + t * 16 + mrow];
#pragma unroll
        for (int i = 0; i < 8; ++i) {
            float v = c[i] + bb;                 // row M = half*8+i, col = t*16+mrow
            v = v / (1.0f + __expf(-v));         // SiLU
            hstage[warp][half * 8 + i][t * 16 + mrow] = (f16)v;
        }
    }
    __syncthreads();

    // ---- GEMM2 A fragments from LDS (two K-halves of 32) ----
    v16h a2[2];
#pragma unroll
    for (int kf = 0; kf < 2; ++kf) {
        const f16* hp = &hstage[warp][mrow][kf * 32 + half * 8];
#pragma unroll
        for (int j = 0; j < 8; ++j) {
            a2[kf][j]     = hp[j];
            a2[kf][8 + j] = hp[16 + j];
        }
    }

    // ---- GEMM2: w = h @ w2 + b2 -> 12 col tiles of 16, staged to LDS f32 ----
#pragma unroll
    for (int t = 0; t < 12; ++t) {
        v8f acc = {};
#pragma unroll
        for (int kf = 0; kf < 2; ++kf) {
            v16h bf = *(const v16h*)(w2f + ((((size_t)layer * 12 + t) * 2 + kf) * 32 + lane) * 16);
            acc = __builtin_amdgcn_wmma_f32_16x16x32_f16(false, a2[kf], false, bf,
                                                         (short)0, acc, false, false);
        }
        int col = t * 16 + mrow;
        float bb = b2[layer * W2OUT + col];
#pragma unroll
        for (int i = 0; i < 8; ++i)
            wstage[warp][half * 8 + i][col] = acc[i] + bb;
    }
    __syncthreads();

    // ---- Message phase: lane == channel, loop over the wave's 16 edges ----
    for (int el = 0; el < 16; ++el) {
        int e = tileBase + el;
        int s = snd[e], r = rcv[e];
        float co = cutoff[e] * INV_AVG;
        const float* yv = Y + (size_t)e * MMC;
        const float* w6 = &wstage[warp][el][lane * 6];  // [c][path][l] -> c*6+p*3+l
        float wa0 = w6[0], wa1 = w6[1], wa2 = w6[2];    // self path, l=0,1,2
        float wb0 = w6[3], wb1 = w6[4], wb2 = w6[5];    // SH path,   l=0,1,2
        const float* xs = feats + ((size_t)s * CC + lane) * MMC;
        float*       ag = agg   + ((size_t)r * CC + lane) * MMC;

        atomicAdd(&ag[0], (wa0 * xs[0] + wb0 * yv[0]) * co);
#pragma unroll
        for (int m = 1; m < 4; ++m)
            atomicAdd(&ag[m], (wa1 * xs[m] + wb1 * yv[m]) * co);
#pragma unroll
        for (int m = 4; m < 9; ++m)
            atomicAdd(&ag[m], (wa2 * xs[m] + wb2 * yv[m]) * co);
    }
}

// ---------------------------------------------------------------------------
// Node update (one per layer): channel mix per l, gate nonlinearity, skip.
// One node per wave; mix weights + agg row staged in LDS. Last layer writes
// alpha-scaled result straight to d_out.
// ---------------------------------------------------------------------------
__global__ __launch_bounds__(256)
void node_kernel(int layer, int isFinal,
                 const float* __restrict__ agg,
                 const float* __restrict__ mixw,
                 const float* __restrict__ featsIn,
                 float* __restrict__ featsOut,
                 float* __restrict__ outFinal)
{
    __shared__ float smix[3 * CC * CC];      // 12 KB: this layer's mix weights
    __shared__ float sagg[8][CC][MMC];       //  9 KB: one agg row per wave

    for (int idx = threadIdx.x; idx < 3 * CC * CC; idx += 256)
        smix[idx] = mixw[(size_t)layer * 3 * CC * CC + idx];
    __syncthreads();

    const int warp = threadIdx.x >> 5;
    const int lane = threadIdx.x & 31;
    const int n = blockIdx.x * 8 + warp;

    const float* ap = agg + ((size_t)n * CC + lane) * MMC;
#pragma unroll
    for (int m = 0; m < MMC; ++m) sagg[warp][lane][m] = ap[m];
    __syncthreads();

    const int d = lane;                       // output channel
    float mixed[MMC];
#pragma unroll
    for (int m = 0; m < MMC; ++m) {
        int l = (m == 0) ? 0 : ((m < 4) ? 1 : 2);
        const float* mw = &smix[l * CC * CC];
        float acc = 0.0f;
#pragma unroll
        for (int c = 0; c < CC; ++c)
            acc += sagg[warp][c][m] * mw[c * CC + d];
        mixed[m] = acc;
    }

    float s  = mixed[0];
    float sg = 1.0f / (1.0f + __expf(-s));    // sigmoid(s)
    float out[MMC];
    out[0] = s * sg;                          // silu(s)
#pragma unroll
    for (int m = 1; m < MMC; ++m) out[m] = sg * mixed[m];

    if (layer > 0) {
        const float* fi = featsIn + ((size_t)n * CC + d) * MMC;
#pragma unroll
        for (int m = 0; m < MMC; ++m) out[m] += fi[m];
    }

    if (isFinal) {
        float* o = outFinal + ((size_t)n * CC + d) * MMC;
        o[0] = out[0];
#pragma unroll
        for (int m = 1; m < 4; ++m) o[m] = 0.5f  * out[m];
#pragma unroll
        for (int m = 4; m < 9; ++m) o[m] = 0.25f * out[m];
    } else {
        float* fo = featsOut + ((size_t)n * CC + d) * MMC;
#pragma unroll
        for (int m = 0; m < MMC; ++m) fo[m] = out[m];
    }
}

// ---------------------------------------------------------------------------
// Host driver
// ---------------------------------------------------------------------------
extern "C" void kernel_launch(void* const* d_in, const int* in_sizes, int n_in,
                              void* d_out, int out_size, void* d_ws, size_t ws_size,
                              hipStream_t stream)
{
    (void)in_sizes; (void)n_in; (void)out_size; (void)ws_size;

    const float* positions = (const float*)d_in[0];
    const float* embed     = (const float*)d_in[1];
    const float* mlp_w1    = (const float*)d_in[2];
    const float* mlp_b1    = (const float*)d_in[3];
    const float* mlp_w2    = (const float*)d_in[4];
    const float* mlp_b2    = (const float*)d_in[5];
    const float* mix_w     = (const float*)d_in[6];
    const int*   species   = (const int*)d_in[7];
    const int*   senders   = (const int*)d_in[8];
    const int*   receivers = (const int*)d_in[9];
    float*       out       = (float*)d_out;

    char*  ws  = (char*)d_ws;
    size_t off = 0;
    auto alloc = [&](size_t bytes) -> char* {
        char* p = ws + off;
        off = (off + bytes + 255) & ~(size_t)255;
        return p;
    };

    float* Yp     = (float*)alloc((size_t)EE * MMC * sizeof(float));
    float* cutp   = (float*)alloc((size_t)EE * sizeof(float));
    f16*   bessp  = (f16*)  alloc((size_t)EE * NBQ * sizeof(f16));
    f16*   w1fp   = (f16*)  alloc((size_t)NLAYERS * 4 * 32 * 16 * sizeof(f16));
    f16*   w2fp   = (f16*)  alloc((size_t)NLAYERS * 12 * 2 * 32 * 16 * sizeof(f16));
    float* aggp   = (float*)alloc((size_t)NN * CC * MMC * sizeof(float));
    float* featsA = (float*)alloc((size_t)NN * CC * MMC * sizeof(float));
    float* featsB = (float*)alloc((size_t)NN * CC * MMC * sizeof(float));

    // one-time (per launch) prep
    init_feats<<<(NN * CC) / 256, 256, 0, stream>>>(embed, species, featsA);
    geom_kernel<<<EE / 256, 256, 0, stream>>>(positions, senders, receivers,
                                              Yp, cutp, bessp);
    conv_w1<<<(NLAYERS * 4 * 32 * 16) / 256, 256, 0, stream>>>(mlp_w1, w1fp);
    conv_w2<<<(NLAYERS * 12 * 2 * 32 * 16) / 256, 256, 0, stream>>>(mlp_w2, w2fp);

    float* fcur  = featsA;
    float* fnext = featsB;
    for (int layer = 0; layer < NLAYERS; ++layer) {
        zero_kernel<<<(NN * CC * MMC) / 256, 256, 0, stream>>>(aggp, NN * CC * MMC);
        edge_kernel<<<EE / 64, 128, 0, stream>>>(layer, bessp, Yp, cutp, fcur,
                                                 w1fp, w2fp, mlp_b1, mlp_b2,
                                                 senders, receivers, aggp);
        int isFinal = (layer == NLAYERS - 1);
        node_kernel<<<NN / 8, 256, 0, stream>>>(layer, isFinal, aggp, mix_w,
                                                fcur, fnext, out);
        float* tmp = fcur; fcur = fnext; fnext = tmp;
    }
}
